// Attention_42202348650566
// MI455X (gfx1250) — compile-verified, matrix operand order checked by reading, not run on hardware
//
#include <hip/hip_runtime.h>

// ---------------------------------------------------------------------------
// Gated attention (AlphaFold-style) for MI455X / gfx1250 (wave32, WMMA).
// B=64, Q=K=512, C=256, H=8, HD=32, OUT=256.
//
// Pipeline:
//   kern_convert : f32 -> bf16 activations, transposed bf16 weights
//   kern_proj    : WMMA GEMMs -> q_proj, k_proj, v^T (bf16), sigmoid gate (f32)
//   kern_attn    : flash attention per wave (16 q rows), WMMA QK^T and PV,
//                  online softmax, gating; writes gated weighted_avg (bf16)
//   kern_out     : WMMA GEMM + output bias -> f32 d_out
// ---------------------------------------------------------------------------

#define BB 64
#define QQ 512
#define KK 512
#define CC 256
#define HH 8
#define HD 32
#define OUT_C 256
#define ROWS (BB * QQ)          // 32768 global rows
#define INV_SQRT_HD 0.17677669529663687f

typedef __attribute__((ext_vector_type(16))) __bf16 v16bf;
typedef __attribute__((ext_vector_type(8)))  float  v8f;

struct bf16x16_bits { uint4 lo, hi; };

__device__ __forceinline__ unsigned short f2bf(float f) {
    __bf16 h = (__bf16)f;
    return __builtin_bit_cast(unsigned short, h);
}

// Load one 16x32 bf16 operand tile (A layout; B uses the mirrored layout with
// N<->M, which is identical addressing when the source is stored so that the
// B-matrix columns are contiguous rows in memory).
// Per ISA 7.12.2 (16-bit A 16x32): lanes 0-15 hold row M=lane with K 0..7 in
// v[0..3] and K 16..23 in v[4..7]; lanes 16-31 hold K 8..15 / 24..31.
__device__ __forceinline__ v16bf load_tile16x32(const unsigned short* base, int stride) {
    const int lane = threadIdx.x & 31;
    const int row  = lane & 15;
    const int half = (lane >> 4) & 1;
    const unsigned short* p = base + row * stride + half * 8;
    bf16x16_bits t;
    t.lo = *(const uint4*)(p);        // K = half*8 .. half*8+7
    t.hi = *(const uint4*)(p + 16);   // K = 16+half*8 .. 16+half*8+7
    return __builtin_bit_cast(v16bf, t);
}

__device__ __forceinline__ v8f wmma_bf16(v16bf a, v16bf b, v8f c) {
    return __builtin_amdgcn_wmma_f32_16x16x32_bf16(
        /*neg_a=*/false, a, /*neg_b=*/false, b,
        /*c_mod=*/(short)0, c, /*reuse_a=*/false, /*reuse_b=*/false);
}

__device__ __forceinline__ float rmax16(float v) {
    for (int off = 1; off < 16; off <<= 1)
        v = fmaxf(v, __shfl_xor(v, off, 32));
    return v;
}
__device__ __forceinline__ float rsum16(float v) {
    for (int off = 1; off < 16; off <<= 1)
        v += __shfl_xor(v, off, 32);
    return v;
}

// ---------------------------------------------------------------------------
// Kernel 0: bf16 conversion + weight transposes.
// ---------------------------------------------------------------------------
__global__ void kern_convert(const float* __restrict__ qd, const float* __restrict__ md,
                             const float* __restrict__ wq, const float* __restrict__ wk,
                             const float* __restrict__ wv, const float* __restrict__ wg,
                             const float* __restrict__ wo,
                             unsigned short* __restrict__ qd_bf, unsigned short* __restrict__ md_bf,
                             unsigned short* __restrict__ wq_t, unsigned short* __restrict__ wk_t,
                             unsigned short* __restrict__ wv_t, unsigned short* __restrict__ wg_t,
                             unsigned short* __restrict__ wo_t) {
    const int tid = blockIdx.x * blockDim.x + threadIdx.x;
    const int nthr = gridDim.x * blockDim.x;
    const int nact = ROWS * CC;   // 8,388,608
    for (int i = tid; i < nact; i += nthr) {
        qd_bf[i] = f2bf(qd[i]);
        md_bf[i] = f2bf(md[i]);
    }
    // weights: [256 in][256 outcol] row-major -> transposed [outcol][in]
    for (int i = tid; i < CC * CC; i += nthr) {
        const int a = i >> 8;       // input dim (contraction)
        const int o = i & 255;      // output col
        const int t = o * CC + a;
        wq_t[t] = f2bf(wq[i] * INV_SQRT_HD);  // fold HD^-0.5 into q projection
        wk_t[t] = f2bf(wk[i]);
        wv_t[t] = f2bf(wv[i]);
        wg_t[t] = f2bf(wg[i]);
        wo_t[t] = f2bf(wo[i]);                // output_w: [(h*HD+hd)][o]
    }
}

// ---------------------------------------------------------------------------
// Kernel 1: fused projections. One wave = one 16x16 output tile of one of the
// four GEMMs (0=query,1=key,2=value,3=gating). 4 * 2048 * 16 = 131072 tiles.
// ---------------------------------------------------------------------------
__global__ void kern_proj(const unsigned short* __restrict__ qd_bf,
                          const unsigned short* __restrict__ md_bf,
                          const unsigned short* __restrict__ wq_t,
                          const unsigned short* __restrict__ wk_t,
                          const unsigned short* __restrict__ wv_t,
                          const unsigned short* __restrict__ wg_t,
                          const float* __restrict__ gating_b,
                          unsigned short* __restrict__ q_proj,   // [b,h,q,hd] bf16
                          unsigned short* __restrict__ k_proj,   // [b,h,k,hd] bf16
                          unsigned short* __restrict__ v_t,      // [b,h,hd,k] bf16
                          float* __restrict__ gate) {            // [b,h,q,hd] f32
    const int wid  = (blockIdx.x * blockDim.x + threadIdx.x) >> 5;
    const int proj = wid >> 15;            // 32768 tiles per projection
    const int rem  = wid & 32767;
    const int mt   = rem >> 4;             // 0..2047  (16-row tiles of 32768)
    const int nt   = rem & 15;             // 0..15    (16-col tiles of 256)

    const unsigned short* A  = (proj == 0 || proj == 3) ? qd_bf : md_bf;
    const unsigned short* Wt = (proj == 0) ? wq_t : (proj == 1) ? wk_t
                             : (proj == 2) ? wv_t : wg_t;

    v8f acc = {};
    const unsigned short* abase = A  + mt * 16 * CC;
    const unsigned short* bbase = Wt + nt * 16 * CC;
    for (int kc = 0; kc < CC; kc += 32) {
        v16bf a = load_tile16x32(abase + kc, CC);
        v16bf b = load_tile16x32(bbase + kc, CC);
        acc = wmma_bf16(a, b, acc);
    }

    const int lane = threadIdx.x & 31;
    const int col  = lane & 15;
    const int h    = nt >> 1;              // 32 cols per head
    const int hd0  = (nt & 1) * 16 + col;  // hd index of this lane's column
    for (int r = 0; r < 8; ++r) {
        const int grow = mt * 16 + ((lane < 16) ? r : 8 + r);  // b*512 + pos
        const int b_   = grow >> 9;
        const int pos  = grow & 511;
        const int bh   = b_ * HH + h;
        if (proj == 0) {
            q_proj[((size_t)(bh * QQ + pos)) * HD + hd0] = f2bf(acc[r]);
        } else if (proj == 1) {
            k_proj[((size_t)(bh * KK + pos)) * HD + hd0] = f2bf(acc[r]);
        } else if (proj == 2) {
            v_t[((size_t)(bh * HD + hd0)) * KK + pos] = f2bf(acc[r]);
        } else {
            const float x = acc[r] + gating_b[h * HD + hd0];
            gate[((size_t)(bh * QQ + pos)) * HD + hd0] = 1.0f / (1.0f + __expf(-x));
        }
    }
}

// ---------------------------------------------------------------------------
// Kernel 2: flash attention. One wave = one (b, h, 16-row q tile).
// 64*8*32 = 16384 wave-tiles; blocks of 8 waves.
// ---------------------------------------------------------------------------
__global__ void kern_attn(const unsigned short* __restrict__ q_proj,
                          const unsigned short* __restrict__ k_proj,
                          const unsigned short* __restrict__ v_t,
                          const float* __restrict__ gate,
                          const float* __restrict__ bias,      // [B,1,Q,K]
                          const float* __restrict__ nb,        // [H,Q,K]
                          unsigned short* __restrict__ wa) {   // [b,q,h*HD] bf16 gated avg
    __shared__ __align__(16) unsigned short p_lds[8 * 16 * 32];

    const int wslot = threadIdx.x >> 5;
    const int wtile = blockIdx.x * 8 + wslot;
    const int qt = wtile & 31;          // q tile (16 rows)
    const int bh = wtile >> 5;          // 0..511
    const int h  = bh & 7;
    const int b  = bh >> 3;

    const int lane = threadIdx.x & 31;
    const int col  = lane & 15;

    // A operand: q tile 16x32 (full head-dim contraction), loaded once.
    const v16bf a_q = load_tile16x32(q_proj + ((size_t)(bh * QQ + qt * 16)) * HD, HD);

    const float* bias_b = bias + (size_t)b * QQ * KK;
    const float* nb_h   = nb   + (size_t)h * QQ * KK;

    float mrow[8], lrow[8];
    v8f o0 = {}, o1 = {};
    for (int r = 0; r < 8; ++r) { mrow[r] = -1e30f; lrow[r] = 0.0f; }

    unsigned short* pl = &p_lds[wslot * 16 * 32];

    for (int kc = 0; kc < KK; kc += 32) {
        // S = q . k^T  (two N=16 tiles cover 32 k positions)
        v16bf bk0 = load_tile16x32(k_proj + ((size_t)(bh * KK + kc)) * HD, HD);
        v16bf bk1 = load_tile16x32(k_proj + ((size_t)(bh * KK + kc + 16)) * HD, HD);
        v8f z = {};
        v8f s0 = wmma_bf16(a_q, bk0, z);
        v8f s1 = wmma_bf16(a_q, bk1, z);

        float p0[8], p1[8];
        for (int r = 0; r < 8; ++r) {
            const int qrow = qt * 16 + ((lane < 16) ? r : 8 + r);
            const int k0 = kc + col;
            const float e0 = s0[r] + bias_b[(size_t)qrow * KK + k0]
                                   + nb_h[(size_t)qrow * KK + k0];
            const float e1 = s1[r] + bias_b[(size_t)qrow * KK + k0 + 16]
                                   + nb_h[(size_t)qrow * KK + k0 + 16];
            const float mnew = fmaxf(mrow[r], rmax16(fmaxf(e0, e1)));
            const float sc = __expf(mrow[r] - mnew);
            mrow[r] = mnew;
            p0[r] = __expf(e0 - mnew);
            p1[r] = __expf(e1 - mnew);
            lrow[r] = lrow[r] * sc + rsum16(p0[r] + p1[r]);
            o0[r] *= sc;
            o1[r] *= sc;
        }

        // P (C/D layout, f32) -> LDS -> A layout (bf16)
        for (int r = 0; r < 8; ++r) {
            const int row = (lane < 16) ? r : 8 + r;
            pl[row * 32 + col]      = f2bf(p0[r]);
            pl[row * 32 + 16 + col] = f2bf(p1[r]);
        }
        asm volatile("s_wait_dscnt 0" ::: "memory");  // wave-private LDS RAW fence
        v16bf a_p = load_tile16x32(pl, 32);

        // O += P . V   (v^T rows are hd, contiguous over k)
        v16bf bv0 = load_tile16x32(v_t + ((size_t)(bh * HD + 0))  * KK + kc, KK);
        v16bf bv1 = load_tile16x32(v_t + ((size_t)(bh * HD + 16)) * KK + kc, KK);
        o0 = wmma_bf16(a_p, bv0, o0);
        o1 = wmma_bf16(a_p, bv1, o1);
    }

    // epilogue: normalize, gate, store gated weighted_avg as bf16 [row, h*HD+hd]
    for (int r = 0; r < 8; ++r) {
        const int qrow = qt * 16 + ((lane < 16) ? r : 8 + r);
        const float inv = 1.0f / lrow[r];
        const size_t gbase = ((size_t)(bh * QQ + qrow)) * HD;
        const float g0 = gate[gbase + col];
        const float g1 = gate[gbase + 16 + col];
        const size_t orow = ((size_t)(b * QQ + qrow)) * CC + h * HD;
        wa[orow + col]      = f2bf(o0[r] * inv * g0);
        wa[orow + 16 + col] = f2bf(o1[r] * inv * g1);
    }
}

// ---------------------------------------------------------------------------
// Kernel 3: output projection [32768,256] x [256,256] + bias -> f32.
// One wave = one 16x16 tile; 2048*16 = 32768 tiles.
// ---------------------------------------------------------------------------
__global__ void kern_out(const unsigned short* __restrict__ wa,
                         const unsigned short* __restrict__ wo_t,
                         const float* __restrict__ out_b,
                         float* __restrict__ out) {
    const int wid = (blockIdx.x * blockDim.x + threadIdx.x) >> 5;
    const int mt = wid >> 4;
    const int nt = wid & 15;

    v8f acc = {};
    const unsigned short* abase = wa   + (size_t)mt * 16 * CC;
    const unsigned short* bbase = wo_t + (size_t)nt * 16 * CC;
    for (int kc = 0; kc < CC; kc += 32) {
        v16bf a = load_tile16x32(abase + kc, CC);
        v16bf b = load_tile16x32(bbase + kc, CC);
        acc = wmma_bf16(a, b, acc);
    }

    const int lane = threadIdx.x & 31;
    const int ncol = nt * 16 + (lane & 15);
    const float ob = out_b[ncol];
    for (int r = 0; r < 8; ++r) {
        const int row = mt * 16 + ((lane < 16) ? r : 8 + r);
        out[(size_t)row * OUT_C + ncol] = acc[r] + ob;
    }
}

// ---------------------------------------------------------------------------
extern "C" void kernel_launch(void* const* d_in, const int* in_sizes, int n_in,
                              void* d_out, int out_size, void* d_ws, size_t ws_size,
                              hipStream_t stream) {
    (void)in_sizes; (void)n_in; (void)out_size; (void)ws_size;

    const float* q_data   = (const float*)d_in[0];
    const float* m_data   = (const float*)d_in[1];
    const float* bias     = (const float*)d_in[2];
    const float* nb       = (const float*)d_in[3];
    const float* query_w  = (const float*)d_in[4];
    const float* key_w    = (const float*)d_in[5];
    const float* value_w  = (const float*)d_in[6];
    const float* gating_w = (const float*)d_in[7];
    const float* gating_b = (const float*)d_in[8];
    const float* output_w = (const float*)d_in[9];
    const float* output_b = (const float*)d_in[10];
    float* out = (float*)d_out;

    // workspace carve-up (all 256B aligned)
    char* w = (char*)d_ws;
    size_t off = 0;
    auto carve = [&](size_t bytes) -> void* {
        void* p = w + off;
        off += (bytes + 255) & ~(size_t)255;
        return p;
    };
    const size_t act_b  = (size_t)ROWS * CC * sizeof(unsigned short);   // 16 MB
    const size_t wt_b   = (size_t)CC * CC * sizeof(unsigned short);     // 128 KB
    const size_t proj_b = (size_t)BB * HH * QQ * HD * sizeof(unsigned short); // 16 MB

    unsigned short* qd_bf = (unsigned short*)carve(act_b);
    unsigned short* md_bf = (unsigned short*)carve(act_b);
    unsigned short* wq_t  = (unsigned short*)carve(wt_b);
    unsigned short* wk_t  = (unsigned short*)carve(wt_b);
    unsigned short* wv_t  = (unsigned short*)carve(wt_b);
    unsigned short* wg_t  = (unsigned short*)carve(wt_b);
    unsigned short* wo_t  = (unsigned short*)carve(wt_b);
    unsigned short* qp_ws = (unsigned short*)carve(proj_b);
    unsigned short* kp_ws = (unsigned short*)carve(proj_b);
    unsigned short* vt_ws = (unsigned short*)carve(proj_b);
    float*          gate_ws = (float*)carve((size_t)BB * HH * QQ * HD * sizeof(float));
    unsigned short* wa_ws   = (unsigned short*)carve(act_b);

    kern_convert<<<2048, 256, 0, stream>>>(q_data, m_data, query_w, key_w, value_w,
                                           gating_w, output_w,
                                           qd_bf, md_bf, wq_t, wk_t, wv_t, wg_t, wo_t);

    // 4 projections * 32768 tiles = 131072 waves / 8 waves per block
    kern_proj<<<16384, 256, 0, stream>>>(qd_bf, md_bf, wq_t, wk_t, wv_t, wg_t,
                                         gating_b, qp_ws, kp_ws, vt_ws, gate_ws);

    // 16384 wave-tiles / 8 per block
    kern_attn<<<2048, 256, 0, stream>>>(qp_ws, kp_ws, vt_ws, gate_ws, bias, nb, wa_ws);

    // 32768 wave-tiles / 8 per block
    kern_out<<<4096, 256, 0, stream>>>(wa_ws, wo_t, output_b, out);
}
